// ESM_GATNet_8108898255302
// MI455X (gfx1250) — matrix operand order repaired and checked by reading, not compiled, not run on hardware
//
#include <hip/hip_runtime.h>
#include <math.h>

typedef __attribute__((ext_vector_type(16))) _Float16 v16h;
typedef __attribute__((ext_vector_type(8)))  float    v8f;

#define THREADS 256

namespace {
constexpr int NN   = 100000;   // nodes
constexpr int EIN  = 400000;   // input edges
constexpr int ETOT = 500000;   // + self loops
constexpr int NG   = 2048;     // graphs
constexpr int FIN  = 78;       // per-head dim layer 1
constexpr int H1N  = 10;       // heads layer 1
constexpr int D1   = 780;      // H1N*FIN
constexpr int K1P  = 96;       // FIN padded to mult of 32
constexpr int D1P  = 800;      // D1 padded to mult of 32 (K of gemm2)
constexpr int C2   = 128;      // layer-2 out dim
constexpr int DT   = 1280;     // target dim
constexpr int DXT  = 256;      // target projection
constexpr int DXC  = 512;      // concat dim
constexpr int DF1  = 1024;
constexpr int DF2  = 256;
}

// ---------------- device helpers ----------------

__device__ __forceinline__ void atomicMaxF(float* addr, float val) {
  // order-preserving int trick; buffer initialized to -inf (0xFF800000)
  if (val >= 0.0f) atomicMax((int*)addr, __float_as_int(val));
  else             atomicMin((unsigned int*)addr, __float_as_uint(val));
}

// ---------------- WMMA GEMM ----------------
// C[M,N] = A[M,Kp](f16 row-major) x B(f16, fragment-packed) (+bias)(+relu)
// One wave owns a 16 x (16*NTW) strip of C: A fragment loaded once per K-step,
// NTW WMMAs issued against NTW packed-B fragments (each one contiguous 32B/lane).
// A fragment (16x32): lane lr holds row, half i -> k = i + 16*lanehi
// B packed: tile (kt,nt) at (kt*ntiles+nt)*512 + lane*16 + i  (i -> k = i+16*lanehi, lr -> col)
// D (16x16 f32): reg r -> row = r + 8*lanehi, lane lr -> col
template <int NTW>
__global__ void wmma_gemm_f16f32(const _Float16* __restrict__ A,
                                 const _Float16* __restrict__ Bp,
                                 float* __restrict__ C,
                                 const float* __restrict__ bias,
                                 int M, int N, int Kp, int lda, int ldc,
                                 int ntiles, int total_waves, int act) {
  int wave = (int)((blockIdx.x * blockDim.x + threadIdx.x) >> 5);
  if (wave >= total_waves) return;            // whole-wave exit: EXEC stays all-1s
  int lane    = threadIdx.x & 31;
  int ngroups = ntiles / NTW;
  int mt  = wave / ngroups;
  int g   = wave - mt * ngroups;
  int nt0 = g * NTW;
  int lanehi = lane >> 4;
  int lr     = lane & 15;

  const _Float16* Arow  = A + (size_t)(mt * 16 + lr) * lda;
  const _Float16* Bbase = Bp + (size_t)nt0 * 512 + lane * 16;

  v8f acc[NTW];
#pragma unroll
  for (int j = 0; j < NTW; ++j) acc[j] = v8f{};

  for (int k0 = 0; k0 < Kp; k0 += 32) {
    v16h a;
    const _Float16* Ap = Arow + k0;
#pragma unroll
    for (int i = 0; i < 16; ++i)
      a[i] = Ap[i + 16 * lanehi];
    const _Float16* Bk = Bbase + (size_t)(k0 >> 5) * ntiles * 512;
#pragma unroll
    for (int j = 0; j < NTW; ++j) {
      v16h b = *(const v16h*)(Bk + (size_t)j * 512);   // 32B contiguous, aligned
      acc[j] = __builtin_amdgcn_wmma_f32_16x16x32_f16(false, a, false, b,
                                                      (short)0, acc[j], false, false);
    }
  }

  int rbase = mt * 16 + lanehi * 8;
#pragma unroll
  for (int j = 0; j < NTW; ++j) {
    int col = (nt0 + j) * 16 + lr;
    if (col < N) {
      float bv = bias ? bias[col] : 0.0f;
#pragma unroll
      for (int r = 0; r < 8; ++r) {
        float v = acc[j][r] + bv;
        if (act) v = v > 0.0f ? v : 0.0f;
        C[(size_t)(rbase + r) * ldc + col] = v;
      }
    }
  }
}

// ---------------- elementwise / conversion kernels ----------------

__global__ void fill_f32(float* __restrict__ p, float v, long long n) {
  long long t = (long long)blockIdx.x * blockDim.x + threadIdx.x;
  if (t < n) p[t] = v;
}

// f32 [M,K] -> f16 [M,Kpad], zero pad K..Kpad-1
__global__ void cvt_rows_f16(const float* __restrict__ src, _Float16* __restrict__ dst,
                             int M, int K, int Kpad) {
  long long t = (long long)blockIdx.x * blockDim.x + threadIdx.x;
  long long total = (long long)M * Kpad;
  if (t >= total) return;
  int kp = (int)(t % Kpad);
  long long m = t / Kpad;
  float v = (kp < K) ? src[m * K + kp] : 0.0f;
  dst[t] = (_Float16)v;
}

// weight f32 [K,N] -> f16 fragment-packed [Kp/32, ntiles, 512], zero padded
__global__ void pack_w_f16(const float* __restrict__ src, _Float16* __restrict__ dst,
                           int K, int N, int Kp, int ntiles) {
  long long t = (long long)blockIdx.x * blockDim.x + threadIdx.x;
  long long total = (long long)(Kp >> 5) * ntiles * 512;
  if (t >= total) return;
  int q = (int)(t & 511);
  long long tile = t >> 9;
  int nt = (int)(tile % ntiles);
  int kt = (int)(tile / ntiles);
  int lane = q >> 4, i = q & 15;
  int lanehi = lane >> 4, lr = lane & 15;
  int k = kt * 32 + i + 16 * lanehi;
  int n = nt * 16 + lr;
  float v = (k < K && n < N) ? src[(size_t)k * N + n] : 0.0f;
  dst[t] = (_Float16)v;
}

// h16[n, 0..D1P) = elu(out1[n,c] + b1[c]) for c<D1 else 0  (f16, layer-2 GEMM input)
__global__ void elu_bias_cvt(const float* __restrict__ out1, const float* __restrict__ b1,
                             _Float16* __restrict__ h16) {
  long long t = (long long)blockIdx.x * blockDim.x + threadIdx.x;
  long long total = (long long)NN * D1P;
  if (t >= total) return;
  int c = (int)(t % D1P);
  long long n = t / D1P;
  float v = 0.0f;
  if (c < D1) {
    v = out1[n * D1 + c] + b1[c];
    v = (v > 0.0f) ? v : expm1f(v);
  }
  h16[t] = (_Float16)v;
}

// per (node, head): alpha_src/alpha_dst dots over per-head features
__global__ void alpha_kernel(const float* __restrict__ hfeat,
                             const float* __restrict__ a_src, const float* __restrict__ a_dst,
                             float* __restrict__ asrc, float* __restrict__ adst,
                             int H, int D, int F) {
  long long t = (long long)blockIdx.x * blockDim.x + threadIdx.x;
  if (t >= (long long)NN * H) return;
  int h = (int)(t % H);
  long long n = t / H;
  const float* hp = hfeat + n * D + h * F;
  const float* as = a_src + h * F;
  const float* ad = a_dst + h * F;
  float s1 = 0.0f, s2 = 0.0f;
  for (int d = 0; d < F; ++d) {
    float v = hp[d];
    s1 += v * as[d];
    s2 += v * ad[d];
  }
  asrc[t] = s1;
  adst[t] = s2;
}

__device__ __forceinline__ void edge_endpoints(const int* __restrict__ ei, int e,
                                               int& s, int& d) {
  if (e < EIN) { s = ei[e]; d = ei[EIN + e]; }
  else         { s = d = e - EIN; }          // self loop
}

// pass 1: segment max of leaky_relu logits per (dst, head)
__global__ void att_max(const float* __restrict__ asrc, const float* __restrict__ adst,
                        const int* __restrict__ ei, float* __restrict__ maxbuf, int H) {
  long long t = (long long)blockIdx.x * blockDim.x + threadIdx.x;
  if (t >= (long long)ETOT * H) return;
  int h = (int)(t % H);
  int e = (int)(t / H);
  int s, d; edge_endpoints(ei, e, s, d);
  float lg = asrc[(size_t)s * H + h] + adst[(size_t)d * H + h];
  lg = (lg >= 0.0f) ? lg : 0.2f * lg;
  atomicMaxF(&maxbuf[(size_t)d * H + h], lg);
}

// pass 2: ex = exp(logit - max[dst]); store per-edge, segment-sum per (dst, head)
__global__ void att_exp(const float* __restrict__ asrc, const float* __restrict__ adst,
                        const int* __restrict__ ei, const float* __restrict__ maxbuf,
                        float* __restrict__ ex, float* __restrict__ sumbuf, int H) {
  long long t = (long long)blockIdx.x * blockDim.x + threadIdx.x;
  if (t >= (long long)ETOT * H) return;
  int h = (int)(t % H);
  int e = (int)(t / H);
  int s, d; edge_endpoints(ei, e, s, d);
  float lg = asrc[(size_t)s * H + h] + adst[(size_t)d * H + h];
  lg = (lg >= 0.0f) ? lg : 0.2f * lg;
  float v = expf(lg - maxbuf[(size_t)d * H + h]);
  ex[t] = v;
  atomicAdd(&sumbuf[(size_t)d * H + h], v);
}

// pass 3: out[dst,:] += h[src,:] * alpha  (one wave per edge, lanes stride features)
__global__ void msg_agg(const float* __restrict__ hfeat, const float* __restrict__ ex,
                        const float* __restrict__ sumbuf, const int* __restrict__ ei,
                        float* __restrict__ out, int H, int D, int F) {
  int wave = (int)((blockIdx.x * blockDim.x + threadIdx.x) >> 5);
  int lane = threadIdx.x & 31;
  if (wave >= ETOT) return;
  int s, d; edge_endpoints(ei, wave, s, d);
  const float* hs = hfeat + (size_t)s * D;
  float* od = out + (size_t)d * D;
  for (int c = lane; c < D; c += 32) {
    int h = c / F;
    float alpha = ex[(size_t)wave * H + h] / (sumbuf[(size_t)d * H + h] + 1e-16f);
    atomicAdd(&od[c], hs[c] * alpha);
  }
}

// in-place: buf = relu(buf + b[c])
__global__ void bias_relu_inplace(float* __restrict__ buf, const float* __restrict__ b,
                                  long long M, int C) {
  long long t = (long long)blockIdx.x * blockDim.x + threadIdx.x;
  if (t >= M * C) return;
  int c = (int)(t % C);
  float v = buf[t] + b[c];
  buf[t] = (v > 0.0f) ? v : 0.0f;
}

// one wave per node: gate = sigmoid(h . w_gate + b_gate), coalesced + shuffle reduce
__global__ void gate_kernel(const float* __restrict__ hfin, const float* __restrict__ wg,
                            const float* __restrict__ bg, float* __restrict__ gate) {
  int wave = (int)((blockIdx.x * blockDim.x + threadIdx.x) >> 5);
  int lane = threadIdx.x & 31;
  if (wave >= NN) return;
  const float* hp = hfin + (size_t)wave * C2;
  float s = 0.0f;
  for (int c = lane; c < C2; c += 32) s += hp[c] * wg[c];
#pragma unroll
  for (int off = 16; off > 0; off >>= 1) s += __shfl_down(s, off, 32);
  if (lane == 0) gate[wave] = 1.0f / (1.0f + expf(-(s + bg[0])));
}

__global__ void readout_kernel(const float* __restrict__ hfin, const float* __restrict__ gate,
                               const int* __restrict__ batch,
                               float* __restrict__ hsum, float* __restrict__ hmax) {
  long long t = (long long)blockIdx.x * blockDim.x + threadIdx.x;
  if (t >= (long long)NN * C2) return;
  int c = (int)(t % C2);
  long long n = t / C2;
  int g = batch[n];
  float v = hfin[t];
  atomicAdd(&hsum[(size_t)g * C2 + c], v * gate[n]);
  atomicMaxF(&hmax[(size_t)g * C2 + c], v);
}

// xc16[g, 0:128]=hsum, [128:256]=hmax, [256:512]=xt   (f16 for fc1 GEMM)
__global__ void concat_xc(const float* __restrict__ hsum, const float* __restrict__ hmax,
                          const float* __restrict__ xt, _Float16* __restrict__ xc) {
  long long t = (long long)blockIdx.x * blockDim.x + threadIdx.x;
  if (t >= (long long)NG * DXC) return;
  int c = (int)(t % DXC);
  long long g = t / DXC;
  float v;
  if (c < 128)      v = hsum[g * C2 + c];
  else if (c < 256) v = hmax[g * C2 + (c - 128)];
  else              v = xt[g * DXT + (c - 256)];
  xc[t] = (_Float16)v;
}

// one wave per graph: out = fc2 . W_out + b_out
__global__ void final_kernel(const float* __restrict__ fc2, const float* __restrict__ wout,
                             const float* __restrict__ bout, float* __restrict__ out) {
  int wave = (int)((blockIdx.x * blockDim.x + threadIdx.x) >> 5);
  int lane = threadIdx.x & 31;
  if (wave >= NG) return;
  const float* fp = fc2 + (size_t)wave * DF2;
  float s = 0.0f;
  for (int c = lane; c < DF2; c += 32) s += fp[c] * wout[c];
#pragma unroll
  for (int off = 16; off > 0; off >>= 1) s += __shfl_down(s, off, 32);
  if (lane == 0) out[wave] = s + bout[0];
}

// ---------------- host orchestration ----------------

extern "C" void kernel_launch(void* const* d_in, const int* in_sizes, int n_in,
                              void* d_out, int out_size, void* d_ws, size_t ws_size,
                              hipStream_t stream) {
  (void)in_sizes; (void)n_in; (void)out_size; (void)ws_size;

  const float* x      = (const float*)d_in[0];
  const int*   ei     = (const int*)  d_in[1];
  const int*   batch  = (const int*)  d_in[2];
  const float* target = (const float*)d_in[3];
  const float* W1     = (const float*)d_in[4];
  const float* a_src1 = (const float*)d_in[5];
  const float* a_dst1 = (const float*)d_in[6];
  const float* b1     = (const float*)d_in[7];
  const float* W2     = (const float*)d_in[8];
  const float* a_src2 = (const float*)d_in[9];
  const float* a_dst2 = (const float*)d_in[10];
  const float* b2     = (const float*)d_in[11];
  const float* w_gate = (const float*)d_in[12];
  const float* b_gate = (const float*)d_in[13];
  const float* W_xt   = (const float*)d_in[14];
  const float* b_xt   = (const float*)d_in[15];
  const float* W_fc1  = (const float*)d_in[16];
  const float* b_fc1  = (const float*)d_in[17];
  const float* W_fc2  = (const float*)d_in[18];
  const float* b_fc2  = (const float*)d_in[19];
  const float* W_out  = (const float*)d_in[20];
  const float* b_out  = (const float*)d_in[21];
  float* out = (float*)d_out;

  // ---- workspace carve-out (with big-buffer lifetime reuse) ----
  char* ws = (char*)d_ws;
  size_t cur = 0;
  auto take = [&](size_t bytes) -> char* {
    char* p = ws + cur;
    cur += (bytes + 255) & ~size_t(255);
    return p;
  };
  // packed weight sizes: (Kp/32) * ntiles * 512 f16 elems
  auto packed_elems = [](int Kp, int N) -> size_t {
    return (size_t)(Kp / 32) * ((N + 15) / 16) * 512;
  };
  _Float16* x16    = (_Float16*)take((size_t)NN * K1P * 2);
  _Float16* W1p    = (_Float16*)take(packed_elems(K1P, D1)  * 2);
  _Float16* W2p    = (_Float16*)take(packed_elems(D1P, C2)  * 2);
  _Float16* Wxtp   = (_Float16*)take(packed_elems(DT,  DXT) * 2);
  _Float16* Wfc1p  = (_Float16*)take(packed_elems(DXC, DF1) * 2);
  _Float16* Wfc2p  = (_Float16*)take(packed_elems(DF1, DF2) * 2);
  char* big1 = take((size_t)NN * D1 * 4);  // h1 (f32) -> reused as h16 (f16, NN*D1P)
  char* big2 = take((size_t)NN * D1 * 4);  // out1 (f32) -> reused as h2 + out2
  float*    h1   = (float*)big1;
  _Float16* h16  = (_Float16*)big1;                       // after msg1, h1 dead
  float*    out1 = (float*)big2;
  float*    h2   = (float*)big2;                          // after elu_bias_cvt, out1 dead
  float*    out2 = (float*)(big2 + (size_t)NN * C2 * 4);  // disjoint from h2
  float* asrc1 = (float*)take((size_t)NN * H1N * 4);
  float* adst1 = (float*)take((size_t)NN * H1N * 4);
  float* max1  = (float*)take((size_t)NN * H1N * 4);
  float* sum1  = (float*)take((size_t)NN * H1N * 4);
  float* ex1   = (float*)take((size_t)ETOT * H1N * 4);
  float* asrc2 = (float*)take((size_t)NN * 4);
  float* adst2 = (float*)take((size_t)NN * 4);
  float* max2  = (float*)take((size_t)NN * 4);
  float* sum2  = (float*)take((size_t)NN * 4);
  float* ex2   = (float*)take((size_t)ETOT * 4);
  float* gate  = (float*)take((size_t)NN * 4);
  float* hsum  = (float*)take((size_t)NG * C2 * 4);
  float* hmax  = (float*)take((size_t)NG * C2 * 4);
  _Float16* tgt16 = (_Float16*)take((size_t)NG * DT * 2);
  float*    xt    = (float*)take((size_t)NG * DXT * 4);
  _Float16* xc16  = (_Float16*)take((size_t)NG * DXC * 2);
  float*    fc1   = (float*)take((size_t)NG * DF1 * 4);
  _Float16* fc116 = (_Float16*)take((size_t)NG * DF1 * 2);
  float*    fc2   = (float*)take((size_t)NG * DF2 * 4);

  auto blocks = [](long long n) { return (int)((n + THREADS - 1) / THREADS); };
  auto pack_w = [&](const float* src, _Float16* dst, int K, int N, int Kp) {
    int ntiles = (N + 15) / 16;
    long long total = (long long)(Kp / 32) * ntiles * 512;
    pack_w_f16<<<blocks(total), THREADS, 0, stream>>>(src, dst, K, N, Kp, ntiles);
  };
  auto gemm = [&](int NTW, const _Float16* A, const _Float16* Bp, float* C,
                  const float* bias, int M, int N, int Kp, int lda, int ldc, int act) {
    int mt = M / 16, ntiles = (N + 15) / 16;
    int waves = mt * (ntiles / NTW);
    int blk = (int)(((long long)waves * 32 + THREADS - 1) / THREADS);
    switch (NTW) {
      case 4: wmma_gemm_f16f32<4><<<blk, THREADS, 0, stream>>>(
                  A, Bp, C, bias, M, N, Kp, lda, ldc, ntiles, waves, act); break;
      case 7: wmma_gemm_f16f32<7><<<blk, THREADS, 0, stream>>>(
                  A, Bp, C, bias, M, N, Kp, lda, ldc, ntiles, waves, act); break;
      case 8: wmma_gemm_f16f32<8><<<blk, THREADS, 0, stream>>>(
                  A, Bp, C, bias, M, N, Kp, lda, ldc, ntiles, waves, act); break;
    }
  };
  const float NEG_INF = -__builtin_inff();

  // ===== Layer 1: h1 = x @ W1 (WMMA, 49 n-tiles -> 7 per wave) =====
  cvt_rows_f16<<<blocks((long long)NN * K1P), THREADS, 0, stream>>>(x, x16, NN, FIN, K1P);
  pack_w(W1, W1p, FIN, D1, K1P);
  gemm(7, x16, W1p, h1, nullptr, NN, D1, K1P, K1P, D1, 0);

  // ===== Layer 1 attention + aggregation =====
  alpha_kernel<<<blocks((long long)NN * H1N), THREADS, 0, stream>>>(
      h1, a_src1, a_dst1, asrc1, adst1, H1N, D1, FIN);
  fill_f32<<<blocks((long long)NN * H1N), THREADS, 0, stream>>>(max1, NEG_INF, (long long)NN * H1N);
  fill_f32<<<blocks((long long)NN * H1N), THREADS, 0, stream>>>(sum1, 0.0f, (long long)NN * H1N);
  att_max<<<blocks((long long)ETOT * H1N), THREADS, 0, stream>>>(asrc1, adst1, ei, max1, H1N);
  att_exp<<<blocks((long long)ETOT * H1N), THREADS, 0, stream>>>(asrc1, adst1, ei, max1, ex1, sum1, H1N);
  fill_f32<<<blocks((long long)NN * D1), THREADS, 0, stream>>>(out1, 0.0f, (long long)NN * D1);
  msg_agg<<<blocks((long long)ETOT * 32), THREADS, 0, stream>>>(h1, ex1, sum1, ei, out1, H1N, D1, FIN);

  // ===== ELU(out1 + b1) -> f16 (layer-2 GEMM input); h1 region reused =====
  elu_bias_cvt<<<blocks((long long)NN * D1P), THREADS, 0, stream>>>(out1, b1, h16);

  // ===== Layer 2: h2 = h16 @ W2 (WMMA, all 8 n-tiles in one wave); out1 region reused =====
  pack_w(W2, W2p, D1, C2, D1P);
  gemm(8, h16, W2p, h2, nullptr, NN, C2, D1P, D1P, C2, 0);

  // ===== Layer 2 attention + aggregation (1 head) =====
  alpha_kernel<<<blocks((long long)NN), THREADS, 0, stream>>>(
      h2, a_src2, a_dst2, asrc2, adst2, 1, C2, C2);
  fill_f32<<<blocks((long long)NN), THREADS, 0, stream>>>(max2, NEG_INF, (long long)NN);
  fill_f32<<<blocks((long long)NN), THREADS, 0, stream>>>(sum2, 0.0f, (long long)NN);
  att_max<<<blocks((long long)ETOT), THREADS, 0, stream>>>(asrc2, adst2, ei, max2, 1);
  att_exp<<<blocks((long long)ETOT), THREADS, 0, stream>>>(asrc2, adst2, ei, max2, ex2, sum2, 1);
  fill_f32<<<blocks((long long)NN * C2), THREADS, 0, stream>>>(out2, 0.0f, (long long)NN * C2);
  msg_agg<<<blocks((long long)ETOT * 32), THREADS, 0, stream>>>(h2, ex2, sum2, ei, out2, 1, C2, C2);
  bias_relu_inplace<<<blocks((long long)NN * C2), THREADS, 0, stream>>>(out2, b2, NN, C2);

  // ===== WeightedSumAndMax readout =====
  gate_kernel<<<blocks((long long)NN * 32), THREADS, 0, stream>>>(out2, w_gate, b_gate, gate);
  fill_f32<<<blocks((long long)NG * C2), THREADS, 0, stream>>>(hsum, 0.0f, (long long)NG * C2);
  fill_f32<<<blocks((long long)NG * C2), THREADS, 0, stream>>>(hmax, NEG_INF, (long long)NG * C2);
  readout_kernel<<<blocks((long long)NN * C2), THREADS, 0, stream>>>(out2, gate, batch, hsum, hmax);

  // ===== target projection: xt = target @ W_xt + b_xt (WMMA) =====
  cvt_rows_f16<<<blocks((long long)NG * DT), THREADS, 0, stream>>>(target, tgt16, NG, DT, DT);
  pack_w(W_xt, Wxtp, DT, DXT, DT);
  gemm(4, tgt16, Wxtp, xt, b_xt, NG, DXT, DT, DT, DXT, 0);

  // ===== MLP head =====
  concat_xc<<<blocks((long long)NG * DXC), THREADS, 0, stream>>>(hsum, hmax, xt, xc16);
  pack_w(W_fc1, Wfc1p, DXC, DF1, DXC);
  gemm(4, xc16, Wfc1p, fc1, b_fc1, NG, DF1, DXC, DXC, DF1, 1);   // relu fused
  cvt_rows_f16<<<blocks((long long)NG * DF1), THREADS, 0, stream>>>(fc1, fc116, NG, DF1, DF1);
  pack_w(W_fc2, Wfc2p, DF1, DF2, DF1);
  gemm(4, fc116, Wfc2p, fc2, b_fc2, NG, DF2, DF1, DF1, DF2, 1);  // relu fused
  final_kernel<<<blocks((long long)NG * 32), THREADS, 0, stream>>>(fc2, W_out, b_out, out);
}
